// CasualSelfAttention_12283606468089
// MI455X (gfx1250) — compile-verified, matrix operand order checked by reading, not run on hardware
//
#include <hip/hip_runtime.h>
#include <hip/hip_bf16.h>

typedef __attribute__((ext_vector_type(16))) _Float16 v16h;
typedef __attribute__((ext_vector_type(8)))  _Float16 v8h;
typedef __attribute__((ext_vector_type(8)))  float    v8f;

#define NEG_INF (-__builtin_huge_valf())

// ---------------------------------------------------------------------------
// WMMA helper: D = A(16x32 f16) * B(32x16 f16) + C(16x16 f32)
// ---------------------------------------------------------------------------
__device__ __forceinline__ v8f wmma_f16(v16h a, v16h b, v8f c) {
  return __builtin_amdgcn_wmma_f32_16x16x32_f16(false, a, false, b, (short)0, c,
                                                false, false);
}

// Fragment loader for 16-bit 16x32 A (row-major [16][>=32], stride ld halves)
// and 32x16 B when B^T is stored row-major [16][>=32].
// Layout (ISA 7.12.2): lane l: row = l&15, kb = (l>=16)?8:0,
//   VGPR0-3 hold k = kb..kb+7 (contiguous), VGPR4-7 hold k = kb+16..kb+23.
__device__ __forceinline__ v16h load_frag(const _Float16* __restrict__ base, int ld) {
  const int l  = threadIdx.x & 31;
  const int rm = l & 15;
  const int kb = (l >> 4) << 3;
  const _Float16* p = base + rm * ld + kb;
  union { v16h v; v8h h[2]; } u;
  u.h[0] = *(const v8h*)(p);
  u.h[1] = *(const v8h*)(p + 16);
  return u.v;
}

__device__ __forceinline__ v16h load_frag_lds(const _Float16* base) {
  const int l  = threadIdx.x & 31;
  const int rm = l & 15;
  const int kb = (l >> 4) << 3;
  const _Float16* p = base + rm * 32 + kb;
  union { v16h v; v8h h[2]; } u;
  u.h[0] = *(const v8h*)(p);
  u.h[1] = *(const v8h*)(p + 16);
  return u.v;
}

// Reductions across the 16 lanes that hold one accumulator row (wave32).
__device__ __forceinline__ float rowmax16(float x) {
  #pragma unroll
  for (int off = 1; off < 16; off <<= 1) x = fmaxf(x, __shfl_xor(x, off, 32));
  return x;
}
__device__ __forceinline__ float rowsum16(float x) {
  #pragma unroll
  for (int off = 1; off < 16; off <<= 1) x += __shfl_xor(x, off, 32);
  return x;
}

// ---------------------------------------------------------------------------
// Kernel 0a: f32 -> f16 elementwise
// ---------------------------------------------------------------------------
__global__ void cvt_f16_kernel(const float* __restrict__ in,
                               _Float16* __restrict__ out, int n) {
  int i = blockIdx.x * blockDim.x + threadIdx.x;
  int stride = gridDim.x * blockDim.x;
  for (; i < n; i += stride) out[i] = (_Float16)in[i];
}

// Kernel 0b: f32 [rows][cols] -> f16 transposed [cols][rows]
__global__ void transpose_f16_kernel(const float* __restrict__ in,
                                     _Float16* __restrict__ out,
                                     int rows, int cols) {
  int i = blockIdx.x * blockDim.x + threadIdx.x;
  int stride = gridDim.x * blockDim.x;
  int total = rows * cols;
  for (; i < total; i += stride) {
    int n = i / rows;          // output row  (original column)
    int k = i - n * rows;      // output col  (original row)
    out[i] = (_Float16)in[(size_t)k * cols + n];
  }
}

// ---------------------------------------------------------------------------
// Kernel 1: QKV GEMM.  A=[8192][768] f16, Bt=[2304][768] f16 (w_attn^T).
// Block tile 128x256, 8 waves = 2(M) x 4(N), wave tile 64x64.
// Per K-step of 32: one clause of 16 global_load_b128 (4 A + 4 B fragments,
// all preloaded into arrays) feeding 16 WMMAs with staggered loadcnt waits.
// Scatters Q,K as [B,H,T,64] f16 and V transposed as [B,H,64,T] f16.
// ---------------------------------------------------------------------------
__global__ void __launch_bounds__(256)
gemm_qkv_kernel(const _Float16* __restrict__ A, const _Float16* __restrict__ Bt,
                const float* __restrict__ bias,
                _Float16* __restrict__ Q, _Float16* __restrict__ K,
                _Float16* __restrict__ VT) {
  const int lane  = threadIdx.x & 31;
  const int wave  = threadIdx.x >> 5;
  const int wm    = wave >> 2;                    // 0..1
  const int wn    = wave & 3;                     // 0..3
  const int row0  = blockIdx.x * 128 + wm * 64;   // M base
  const int col0  = blockIdx.y * 256 + wn * 64;   // N base
  const int nlane = lane & 15;
  const int mhalf = (lane >> 4) << 3;

  v8f acc[4][4];
  #pragma unroll
  for (int i = 0; i < 4; ++i)
    #pragma unroll
    for (int j = 0; j < 4; ++j) acc[i][j] = (v8f){};

  for (int k0 = 0; k0 < 768; k0 += 32) {
    v16h a[4], bf[4];
    #pragma unroll
    for (int i = 0; i < 4; ++i)
      a[i] = load_frag(A + (size_t)(row0 + i * 16) * 768 + k0, 768);
    #pragma unroll
    for (int j = 0; j < 4; ++j)
      bf[j] = load_frag(Bt + (size_t)(col0 + j * 16) * 768 + k0, 768);
    if (k0 + 32 < 768) {  // hint next K-step A into cache (global_prefetch_b8)
      __builtin_prefetch(A + (size_t)(row0 + (lane & 63)) * 768 + k0 + 32, 0, 3);
    }
    #pragma unroll
    for (int j = 0; j < 4; ++j)
      #pragma unroll
      for (int i = 0; i < 4; ++i)
        acc[i][j] = wmma_f16(a[i], bf[j], acc[i][j]);
  }

  #pragma unroll
  for (int j = 0; j < 4; ++j) {
    const int   ng = col0 + j * 16 + nlane;   // global column 0..2303
    const float bv = bias[ng];
    const int sect = ng / 768;                // 0=q 1=k 2=v (uniform per tile)
    const int cc   = ng - sect * 768;
    const int hh   = cc >> 6;                 // head (uniform per tile)
    const int dd   = cc & 63;
    #pragma unroll
    for (int i = 0; i < 4; ++i) {
      const int rbase = row0 + i * 16 + mhalf;
      #pragma unroll
      for (int r = 0; r < 8; ++r) {
        const int rw = rbase + r;             // global row 0..8191
        const int bb = rw >> 10;
        const int tt = rw & 1023;
        const _Float16 val = (_Float16)(acc[i][j][r] + bv);
        const size_t bh = (size_t)(bb * 12 + hh);
        if (sect == 0)      Q [(bh * 1024 + tt) * 64 + dd] = val;
        else if (sect == 1) K [(bh * 1024 + tt) * 64 + dd] = val;
        else                VT[(bh * 64 + dd) * 1024 + tt] = val;
      }
    }
  }
}

// ---------------------------------------------------------------------------
// Kernel 2: causal flash attention.
// Q,K: [B*H][T][64] f16, VT: [B*H][64][T] f16 -> Y: [B][T][768] f16.
// Block = 8 waves; each wave owns a 32-row query tile (2x16 subtiles sharing
// every K/V fragment); K-loop step = 32 keys.
// ---------------------------------------------------------------------------
__global__ void __launch_bounds__(256)
attn_kernel(const _Float16* __restrict__ Q, const _Float16* __restrict__ K,
            const _Float16* __restrict__ VT, _Float16* __restrict__ Y) {
  __shared__ alignas(16) _Float16 ldsP[8][32][32];   // per-wave P tiles (16 KB)

  const int lane  = threadIdx.x & 31;
  const int wave  = threadIdx.x >> 5;
  const int bh    = blockIdx.x;                       // 0..95
  const int qbase = blockIdx.y * 256 + wave * 32;
  const int b     = bh / 12;
  const int h     = bh - b * 12;
  const int nlane = lane & 15;
  const int mhalf = (lane >> 4) << 3;

  const _Float16* Qb = Q  + ((size_t)bh * 1024 + qbase) * 64;
  const _Float16* Kb = K  +  (size_t)bh * 1024 * 64;
  const _Float16* Vb = VT +  (size_t)bh * 64 * 1024;

  v16h qa[2][2];
  #pragma unroll
  for (int t = 0; t < 2; ++t) {
    qa[t][0] = load_frag(Qb + (size_t)t * 16 * 64, 64);        // d = 0..31
    qa[t][1] = load_frag(Qb + (size_t)t * 16 * 64 + 32, 64);   // d = 32..63
  }

  v8f o[2][4];
  float mrow[2][8], lrow[2][8];
  #pragma unroll
  for (int t = 0; t < 2; ++t) {
    #pragma unroll
    for (int c = 0; c < 4; ++c) o[t][c] = (v8f){};
    #pragma unroll
    for (int r = 0; r < 8; ++r) { mrow[t][r] = NEG_INF; lrow[t][r] = 0.0f; }
  }

  const int nsteps = (qbase + 63) >> 5;      // causal clip, 32 keys per step
  for (int s = 0; s < nsteps; ++s) {
    const int jb = s << 5;

    // K fragments, shared by both query subtiles
    const v16h k00 = load_frag(Kb + (size_t)jb * 64, 64);
    const v16h k01 = load_frag(Kb + (size_t)jb * 64 + 32, 64);
    const v16h k10 = load_frag(Kb + (size_t)(jb + 16) * 64, 64);
    const v16h k11 = load_frag(Kb + (size_t)(jb + 16) * 64 + 32, 64);

    #pragma unroll
    for (int t = 0; t < 2; ++t) {
      // scores: S[16x32] = Q[16x64] . K^T[64x32]
      v8f s0 = (v8f){}, s1 = (v8f){};
      s0 = wmma_f16(qa[t][0], k00, s0);
      s0 = wmma_f16(qa[t][1], k01, s0);
      s1 = wmma_f16(qa[t][0], k10, s1);
      s1 = wmma_f16(qa[t][1], k11, s1);

      // online softmax, P tile written to LDS in plain [16][32] layout
      #pragma unroll
      for (int r = 0; r < 8; ++r) {
        const int qrow = qbase + t * 16 + mhalf + r;
        float v0 = s0[r] * 0.125f;                     // 1/sqrt(64)
        float v1 = s1[r] * 0.125f;
        if (jb + nlane      > qrow) v0 = NEG_INF;      // causal mask
        if (jb + 16 + nlane > qrow) v1 = NEG_INF;
        const float mx    = rowmax16(fmaxf(v0, v1));
        const float mnew  = fmaxf(mrow[t][r], mx);
        const float alpha = __expf(mrow[t][r] - mnew);
        const float p0    = __expf(v0 - mnew);
        const float p1    = __expf(v1 - mnew);
        lrow[t][r] = lrow[t][r] * alpha + rowsum16(p0 + p1);
        mrow[t][r] = mnew;
        #pragma unroll
        for (int c = 0; c < 4; ++c) o[t][c][r] = o[t][c][r] * alpha;
        ldsP[wave][t * 16 + mhalf + r][nlane]      = (_Float16)p0;
        ldsP[wave][t * 16 + mhalf + r][16 + nlane] = (_Float16)p1;
      }
    }

    // O[32x64] += P[32x32] . V[32x64]; V fragments shared by both subtiles.
    // Preload all V fragments (one load clause), read P from LDS, then WMMA.
    v16h vf[4];
    #pragma unroll
    for (int c = 0; c < 4; ++c)
      vf[c] = load_frag(Vb + (size_t)(c * 16) * 1024 + jb, 1024);
    const v16h pa0 = load_frag_lds(&ldsP[wave][0][0]);
    const v16h pa1 = load_frag_lds(&ldsP[wave][16][0]);
    #pragma unroll
    for (int c = 0; c < 4; ++c) {
      o[0][c] = wmma_f16(pa0, vf[c], o[0][c]);
      o[1][c] = wmma_f16(pa1, vf[c], o[1][c]);
    }
  }

  // normalize and scatter to Y[B][T][768]
  _Float16* Yb = Y + (size_t)b * 1024 * 768 + (size_t)h * 64;
  #pragma unroll
  for (int t = 0; t < 2; ++t) {
    #pragma unroll
    for (int r = 0; r < 8; ++r) {
      const int   tt  = qbase + t * 16 + mhalf + r;
      const float inv = 1.0f / lrow[t][r];
      #pragma unroll
      for (int c = 0; c < 4; ++c)
        Yb[(size_t)tt * 768 + c * 16 + nlane] = (_Float16)(o[t][c][r] * inv);
    }
  }
}

// ---------------------------------------------------------------------------
// Kernel 3: output projection.  A=Y [8192][768] f16, Bt=w_proj^T [768][768] f16.
// Out = A@B + bias, f32 to d_out.  Same 64x64 wave tile as kernel 1.
// ---------------------------------------------------------------------------
__global__ void __launch_bounds__(256)
gemm_proj_kernel(const _Float16* __restrict__ A, const _Float16* __restrict__ Bt,
                 const float* __restrict__ bias, float* __restrict__ Out) {
  const int lane  = threadIdx.x & 31;
  const int wave  = threadIdx.x >> 5;
  const int wm    = wave >> 2;
  const int wn    = wave & 3;
  const int row0  = blockIdx.x * 128 + wm * 64;
  const int col0  = blockIdx.y * 256 + wn * 64;
  const int nlane = lane & 15;
  const int mhalf = (lane >> 4) << 3;

  v8f acc[4][4];
  #pragma unroll
  for (int i = 0; i < 4; ++i)
    #pragma unroll
    for (int j = 0; j < 4; ++j) acc[i][j] = (v8f){};

  for (int k0 = 0; k0 < 768; k0 += 32) {
    v16h a[4], bf[4];
    #pragma unroll
    for (int i = 0; i < 4; ++i)
      a[i] = load_frag(A + (size_t)(row0 + i * 16) * 768 + k0, 768);
    #pragma unroll
    for (int j = 0; j < 4; ++j)
      bf[j] = load_frag(Bt + (size_t)(col0 + j * 16) * 768 + k0, 768);
    if (k0 + 32 < 768) {
      __builtin_prefetch(A + (size_t)(row0 + (lane & 63)) * 768 + k0 + 32, 0, 3);
    }
    #pragma unroll
    for (int j = 0; j < 4; ++j)
      #pragma unroll
      for (int i = 0; i < 4; ++i)
        acc[i][j] = wmma_f16(a[i], bf[j], acc[i][j]);
  }

  #pragma unroll
  for (int j = 0; j < 4; ++j) {
    const int   ng = col0 + j * 16 + nlane;
    const float bv = bias[ng];
    #pragma unroll
    for (int i = 0; i < 4; ++i) {
      const int rbase = row0 + i * 16 + mhalf;
      #pragma unroll
      for (int r = 0; r < 8; ++r)
        Out[(size_t)(rbase + r) * 768 + ng] = acc[i][j][r] + bv;
    }
  }
}

// ---------------------------------------------------------------------------
// Launch
// ---------------------------------------------------------------------------
extern "C" void kernel_launch(void* const* d_in, const int* in_sizes, int n_in,
                              void* d_out, int out_size, void* d_ws, size_t ws_size,
                              hipStream_t stream) {
  (void)in_sizes; (void)n_in; (void)out_size; (void)ws_size;
  const float* x      = (const float*)d_in[0];   // [8,1024,768]
  const float* w_attn = (const float*)d_in[1];   // [768,2304]
  const float* b_attn = (const float*)d_in[2];   // [2304]
  const float* w_proj = (const float*)d_in[3];   // [768,768]
  const float* b_proj = (const float*)d_in[4];   // [768]
  float* out = (float*)d_out;                    // [8,1024,768]

  const int M = 8 * 1024;       // 8192 rows
  char* ws = (char*)d_ws;
  size_t off = 0;
  auto carve = [&](size_t elems) -> _Float16* {
    _Float16* p = (_Float16*)(ws + off);
    off = (off + elems * sizeof(_Float16) + 255) & ~(size_t)255;
    return p;
  };
  _Float16* xf16 = carve((size_t)M * 768);         // x in f16
  _Float16* waT  = carve((size_t)2304 * 768);      // w_attn^T f16
  _Float16* wpT  = carve((size_t)768 * 768);       // w_proj^T f16
  _Float16* Qb   = carve((size_t)96 * 1024 * 64);  // [B*H][T][64]
  _Float16* Kb   = carve((size_t)96 * 1024 * 64);  // [B*H][T][64]
  _Float16* VTb  = carve((size_t)96 * 64 * 1024);  // [B*H][64][T]
  _Float16* Yb   = carve((size_t)M * 768);         // attention output f16

  // Stage 0: conversions / transposes
  {
    int n = M * 768;
    cvt_f16_kernel<<<(n + 255) / 256, 256, 0, stream>>>(x, xf16, n);
  }
  transpose_f16_kernel<<<(768 * 2304 + 255) / 256, 256, 0, stream>>>(
      w_attn, waT, 768, 2304);
  transpose_f16_kernel<<<(768 * 768 + 255) / 256, 256, 0, stream>>>(
      w_proj, wpT, 768, 768);

  // Stage 1: QKV projection (M=8192, N=2304, K=768), block tile 128x256
  gemm_qkv_kernel<<<dim3(M / 128, 2304 / 256), 256, 0, stream>>>(
      xf16, waT, b_attn, Qb, Kb, VTb);

  // Stage 2: causal attention (96 head-batches x 4 query blocks of 256)
  attn_kernel<<<dim3(96, 4), 256, 0, stream>>>(Qb, Kb, VTb, Yb);

  // Stage 3: output projection (M=8192, N=768, K=768), block tile 128x256
  gemm_proj_kernel<<<dim3(M / 128, 768 / 256), 256, 0, stream>>>(
      Yb, wpT, b_proj, out);
}
